// RPNHead_1202590843573
// MI455X (gfx1250) — compile-verified, matrix operand order checked by reading, not run on hardware
//
#include <hip/hip_runtime.h>
#include <hip/hip_bf16.h>

typedef __bf16 bf16;
typedef __attribute__((ext_vector_type(16))) __bf16 v16bf;
typedef __attribute__((ext_vector_type(8)))  __bf16 v8bf;
typedef __attribute__((ext_vector_type(8)))  float  v8f;
typedef int b128v __attribute__((vector_size(16)));   // matches builtin's V4i

#define B_    8
#define C_    512
#define HW_   4096            // 64*64
#define NPIX  32768           // B_*HW_
#define NA_   9
#define AANCH 36864           // HW_*NA_
#define TM    256
#define TN    128

#if defined(__has_builtin)
#if __has_builtin(__builtin_amdgcn_global_load_async_to_lds_b128)
#define HAVE_ASYNC_LDS 1
#endif
#endif

#define AS1 __attribute__((address_space(1)))
#define AS3 __attribute__((address_space(3)))

static __device__ __forceinline__ AS1 b128v* gptr128(const void* p) {
  return (AS1 b128v*)(unsigned long long)p;   // flat global addr bits
}
static __device__ __forceinline__ AS3 b128v* lptr128(void* p) {
  // generic LDS pointer: addr[31:0] is the LDS offset (aperture rules)
  return (AS3 b128v*)(unsigned int)(unsigned long long)p;
}
static __device__ __forceinline__ void wait_async_lds() {
#if defined(__has_builtin) && __has_builtin(__builtin_amdgcn_s_wait_asynccnt)
  __builtin_amdgcn_s_wait_asynccnt(0);
#else
  asm volatile("s_wait_asynccnt 0x0" ::: "memory");
#endif
}

// ---------------------------------------------------------------------------
// Kernel 1: NCHW fp32 -> NHWC bf16 (tiled transpose through LDS, both coalesced)
// ---------------------------------------------------------------------------
__global__ __launch_bounds__(256) void k_feat_to_bf16(const float* __restrict__ x,
                                                      bf16* __restrict__ xb) {
  __shared__ float tile[64][65];
  int bid = blockIdx.x;
  int sBlk = bid & 63; int t = bid >> 6; int cBlk = t & 7; int b = t >> 3;
  int c0 = cBlk * 64, s0 = sBlk * 64;
  const float* src = x + ((size_t)b * C_ + c0) * HW_ + s0;
  for (int i = threadIdx.x; i < 64 * 64; i += 256) {
    int c = i >> 6, s = i & 63;
    tile[c][s] = src[(size_t)c * HW_ + s];
  }
  __syncthreads();
  bf16* dst = xb + ((size_t)b * HW_ + s0) * C_ + c0;
  for (int i = threadIdx.x; i < 64 * 64; i += 256) {
    int s = i >> 6, c = i & 63;
    dst[(size_t)s * C_ + c] = (bf16)tile[c][s];
  }
}

// ---------------------------------------------------------------------------
// Kernel 2: conv_w (Cout,Cin,3,3) fp32 -> wb[tap][co][ci] bf16
// ---------------------------------------------------------------------------
__global__ __launch_bounds__(256) void k_w_to_bf16(const float* __restrict__ w,
                                                   bf16* __restrict__ wb) {
  size_t i = (size_t)blockIdx.x * 256 + threadIdx.x;  // over 9*512*512
  if (i >= (size_t)9 * 512 * 512) return;
  int ci = (int)(i & 511);
  int rest = (int)(i >> 9);
  int co = rest & 511;
  int tap = rest >> 9;
  wb[i] = (bf16)w[((size_t)co * 512 + ci) * 9 + tap];
}

// ---------------------------------------------------------------------------
// Kernel 3: pack cls_w(18x512) + reg_w(36x512) fp32 -> wcat[64][512] bf16 (pad 0)
// ---------------------------------------------------------------------------
__global__ __launch_bounds__(256) void k_head_w(const float* __restrict__ cls_w,
                                                const float* __restrict__ reg_w,
                                                bf16* __restrict__ wcat) {
  int i = blockIdx.x * 256 + threadIdx.x;  // 64*512
  if (i >= 64 * 512) return;
  int ci = i & 511, n = i >> 9;
  float v = 0.f;
  if (n < 18)      v = cls_w[n * 512 + ci];
  else if (n < 54) v = reg_w[(n - 18) * 512 + ci];
  wcat[i] = (bf16)v;
}

// ---------------------------------------------------------------------------
// Kernel 4: 3x3 conv as implicit GEMM, bf16 WMMA, fused bias+ReLU.
// M = pixels (32768), N = cout (512), K = 9 taps * 512 ci.
// Block tile 256x128, 8 waves (wave32), each wave 4x4 16x16 frags (64x64).
// LDS tiles in WMMA per-lane fragment layout: one 32B ds load per fragment.
// Global->LDS uses GLOBAL_LOAD_ASYNC_TO_LDS_B128 when the builtin exists.
// ---------------------------------------------------------------------------
__global__ __launch_bounds__(256, 1) void k_conv3x3(const bf16* __restrict__ xb,
                                                    const bf16* __restrict__ wb,
                                                    const float* __restrict__ bias,
                                                    bf16* __restrict__ f) {
  __shared__ bf16 lA[2][TM * 32];   // 32 KB
  __shared__ bf16 lB[2][TN * 32];   // 16 KB

  const int bm = blockIdx.x;       // pixel tile 0..127
  const int bn = blockIdx.y;       // cout tile 0..3
  const int tid = threadIdx.x;
  const int lane = tid & 31, wv = tid >> 5;
  const int half = lane >> 4, r = lane & 15;
  const int wm = wv & 3, wn = wv >> 2;   // wave tile: 64 rows x 64 cols
  const int p0 = bm * TM;

  v8f acc[4][4] = {};

  // ---- A loader: this thread owns pixel row `tid`, segs 0..3 -------------
  const int pA = p0 + tid;
  const int pbv = pA >> 12;
  const int rem = pA & 4095;
  const int pyv = rem >> 6, pxv = rem & 63;
  const bf16* aBase = xb + ((size_t)pbv << 21);   // b * 4096 * 512
  int ldsA[4];
#pragma unroll
  for (int q = 0; q < 4; ++q)
    ldsA[q] = (tid >> 4) * 512 + (((q & 1) * 16 + (tid & 15)) * 16) + ((q >> 1) * 8);

  // ---- B loader: chunks u = tid and tid+256 ------------------------------
  const int segB = tid & 3;
  int rowB[2], ldsB[2];
  const bf16* bBase[2];
#pragma unroll
  for (int q = 0; q < 2; ++q) {
    rowB[q] = (tid >> 2) + q * 64;
    ldsB[q] = (rowB[q] >> 4) * 512 + (((segB & 1) * 16 + (rowB[q] & 15)) * 16)
              + ((segB >> 1) * 8);
    bBase[q] = wb + ((size_t)(bn * TN + rowB[q]) << 9) + segB * 8;
  }

  v8bf zero8;
#pragma unroll
  for (int e = 0; e < 8; ++e) zero8[e] = (bf16)0.0f;

  // ---- tap-dependent state (recomputed once per 16 k-iterations) ---------
  int aOff = 0; bool aOk = false; size_t bTapOff = 0;
  auto setTap = [&](int t) {
    int ddy = t / 3 - 1, ddx = t - (t / 3) * 3 - 1;
    int sy = pyv + ddy, sx = pxv + ddx;
    aOk = (sy >= 0) & (sy < 64) & (sx >= 0) & (sx < 64);
    aOff = (sy * 64 + sx) << 9;
    bTapOff = (size_t)t * (512 * 512);
  };

#if HAVE_ASYNC_LDS
  auto issueTile = [&](int buf, int ck) {
#pragma unroll
    for (int q = 0; q < 4; ++q) {
      bf16* dst = &lA[buf][ldsA[q]];
      if (aOk) {
        const bf16* src = aBase + aOff + ck * 32 + q * 8;
        __builtin_amdgcn_global_load_async_to_lds_b128(gptr128(src), lptr128(dst), 0, 0);
      } else {
        *(v8bf*)dst = zero8;
      }
    }
#pragma unroll
    for (int q = 0; q < 2; ++q) {
      const bf16* src = bBase[q] + bTapOff + ck * 32;
      __builtin_amdgcn_global_load_async_to_lds_b128(gptr128(src),
                                                     lptr128(&lB[buf][ldsB[q]]), 0, 0);
    }
  };
#else
  v8bf rA[4], rB[2];
  auto issueLoads = [&](int ck) {
#pragma unroll
    for (int q = 0; q < 4; ++q) {
      rA[q] = zero8;
      if (aOk) rA[q] = *(const v8bf*)(aBase + aOff + ck * 32 + q * 8);
    }
#pragma unroll
    for (int q = 0; q < 2; ++q)
      rB[q] = *(const v8bf*)(bBase[q] + bTapOff + ck * 32);
  };
  auto commit = [&](int buf) {
#pragma unroll
    for (int q = 0; q < 4; ++q) *(v8bf*)(&lA[buf][ldsA[q]]) = rA[q];
#pragma unroll
    for (int q = 0; q < 2; ++q) *(v8bf*)(&lB[buf][ldsB[q]]) = rB[q];
  };
#endif

  auto mma = [&](int buf) {
    v16bf afr[4], bfr[4];
#pragma unroll
    for (int i = 0; i < 4; ++i)
      afr[i] = *(const v16bf*)(&lA[buf][(wm * 4 + i) * 512 + lane * 16]);
#pragma unroll
    for (int j = 0; j < 4; ++j)
      bfr[j] = *(const v16bf*)(&lB[buf][(wn * 4 + j) * 512 + lane * 16]);
#pragma unroll
    for (int i = 0; i < 4; ++i)
#pragma unroll
      for (int j = 0; j < 4; ++j)
        acc[i][j] = __builtin_amdgcn_wmma_f32_16x16x32_bf16(
            false, afr[i], false, bfr[j], (short)0, acc[i][j], false, false);
  };

  const int kiters = 9 * 16;  // 9 taps x 16 ci-chunks of 32
  int tap = 0, ck = 0;
  setTap(0);
#if HAVE_ASYNC_LDS
  issueTile(0, 0);
  wait_async_lds();
#else
  issueLoads(0);
  commit(0);
#endif
  __syncthreads();

  for (int it = 0; it < kiters; ++it) {
    const bool more = (it + 1 < kiters);
    if (more) {
      ++ck;
      if (ck == 16) { ck = 0; ++tap; setTap(tap); }
#if HAVE_ASYNC_LDS
      issueTile((it + 1) & 1, ck);
#else
      issueLoads(ck);
#endif
    }
    mma(it & 1);
#if HAVE_ASYNC_LDS
    if (more) wait_async_lds();
#else
    if (more) commit((it + 1) & 1);
#endif
    __syncthreads();
  }

  // epilogue: bias + relu -> bf16 f[pixel][cout]
#pragma unroll
  for (int i = 0; i < 4; ++i) {
#pragma unroll
    for (int j = 0; j < 4; ++j) {
      int co = bn * TN + (wn * 4 + j) * 16 + r;
      float bv = bias[co];
#pragma unroll
      for (int v = 0; v < 8; ++v) {
        int p = p0 + (wm * 4 + i) * 16 + half * 8 + v;
        float val = acc[i][j][v] + bv;
        val = val > 0.f ? val : 0.f;
        f[((size_t)p << 9) + co] = (bf16)val;
      }
    }
  }
}

// ---------------------------------------------------------------------------
// Kernel 5: 1x1 heads (WMMA, 128 pixels x 64 ch x K=512) + anchor decode.
// ---------------------------------------------------------------------------
__global__ __launch_bounds__(256, 1) void k_head(const bf16* __restrict__ f,
                                                 const bf16* __restrict__ wcat,
                                                 const float* __restrict__ cls_b,
                                                 const float* __restrict__ reg_b,
                                                 const int* __restrict__ img_size,
                                                 float* __restrict__ proposals,
                                                 float* __restrict__ cls_out,
                                                 float* __restrict__ ign) {
  __shared__ float lO[128 * 64];
  const int tid = threadIdx.x, lane = tid & 31, wv = tid >> 5;
  const int half = lane >> 4, r = lane & 15;
  const int wm = wv & 3, wn = wv >> 2;  // 4x2 wave grid; wave: 32 pix x 32 ch
  const int p0 = blockIdx.x * 128;

  v8f acc[2][2] = {};
  for (int kc = 0; kc < 16; ++kc) {
    int k0 = kc * 32;
    v16bf afr[2], bfr[2];
#pragma unroll
    for (int i = 0; i < 2; ++i) {
      int p = p0 + (wm * 2 + i) * 16 + r;
      const bf16* base = f + ((size_t)p << 9) + k0 + half * 8;
      v8bf lo = *(const v8bf*)base;
      v8bf hi = *(const v8bf*)(base + 16);
#pragma unroll
      for (int e = 0; e < 8; ++e) { afr[i][e] = lo[e]; afr[i][e + 8] = hi[e]; }
    }
#pragma unroll
    for (int j = 0; j < 2; ++j) {
      int n = (wn * 2 + j) * 16 + r;
      const bf16* base = wcat + ((size_t)n << 9) + k0 + half * 8;
      v8bf lo = *(const v8bf*)base;
      v8bf hi = *(const v8bf*)(base + 16);
#pragma unroll
      for (int e = 0; e < 8; ++e) { bfr[j][e] = lo[e]; bfr[j][e + 8] = hi[e]; }
    }
#pragma unroll
    for (int i = 0; i < 2; ++i)
#pragma unroll
      for (int j = 0; j < 2; ++j)
        acc[i][j] = __builtin_amdgcn_wmma_f32_16x16x32_bf16(
            false, afr[i], false, bfr[j], (short)0, acc[i][j], false, false);
  }

  // spill (+bias) to LDS: lO[pixel_local][channel]
#pragma unroll
  for (int i = 0; i < 2; ++i) {
#pragma unroll
    for (int j = 0; j < 2; ++j) {
      int n = (wn * 2 + j) * 16 + r;
      float bv = 0.f;
      if (n < 18)      bv = cls_b[n];
      else if (n < 54) bv = reg_b[n - 18];
#pragma unroll
      for (int v = 0; v < 8; ++v) {
        int pl = (wm * 2 + i) * 16 + half * 8 + v;
        lO[pl * 64 + n] = acc[i][j][v] + bv;
      }
    }
  }
  __syncthreads();

  // anchor decode
  const float RQ[3] = {0.70710678118654752f, 1.0f, 1.41421356237309505f};
  const float SC[3] = {128.f, 256.f, 512.f};
  for (int u = tid; u < 128 * NA_; u += 256) {
    int pl = u / NA_, a = u - pl * NA_;
    int P = p0 + pl;
    int b = P >> 12, rem = P & 4095, y = rem >> 6, x = rem & 63;
    float* row = &lO[pl * 64];
    float c0 = row[a * 2 + 0], c1 = row[a * 2 + 1];
    float g0 = row[18 + a * 4 + 0], g1 = row[18 + a * 4 + 1];
    float g2 = row[18 + a * 4 + 2], g3 = row[18 + a * 4 + 3];
    int ri = a / 3, si = a - ri * 3;
    float sr = RQ[ri], s = SC[si];
    float aw = s / sr, ah = s * sr;
    float ax = 8.f + x * 16.f, ay = 8.f + y * 16.f;
    float cx = ax + g0 * aw, cy = ay + g1 * ah;
    float pw = aw * __expf(g2), ph = ah * __expf(g3);
    float imw = (float)img_size[b * 2 + 0];
    float imh = (float)img_size[b * 2 + 1];
    float x1 = fminf(fmaxf(cx - pw * 0.5f, 0.f), imw);
    float y1 = fminf(fmaxf(cy - ph * 0.5f, 0.f), imh);
    float x2 = fminf(fmaxf(cx + pw * 0.5f, 0.f), imw);
    float y2 = fminf(fmaxf(cy + ph * 0.5f, 0.f), imh);
    size_t o = (size_t)b * AANCH + (size_t)(y * 64 + x) * NA_ + a;
    float4 pr; pr.x = x1; pr.y = y1; pr.z = x2; pr.w = y2;
    *(float4*)(proposals + o * 4) = pr;
    cls_out[o * 2 + 0] = c0;
    cls_out[o * 2 + 1] = c1;
    ign[o] = 0.0f;  // int32 zero == float 0.0 bit pattern
  }
}

// ---------------------------------------------------------------------------
extern "C" void kernel_launch(void* const* d_in, const int* in_sizes, int n_in,
                              void* d_out, int out_size, void* d_ws, size_t ws_size,
                              hipStream_t stream) {
  const float* feat   = (const float*)d_in[0];
  const int*   img    = (const int*)d_in[1];
  const float* conv_w = (const float*)d_in[2];
  const float* conv_b = (const float*)d_in[3];
  const float* cls_w  = (const float*)d_in[4];
  const float* cls_b  = (const float*)d_in[5];
  const float* reg_w  = (const float*)d_in[6];
  const float* reg_b  = (const float*)d_in[7];

  char* ws = (char*)d_ws;
  const size_t xbBytes = (size_t)NPIX * C_ * 2;           // 33.5 MB
  const size_t wbBytes = (size_t)9 * 512 * 512 * 2;       // 4.7 MB
  const size_t fBytes  = (size_t)NPIX * C_ * 2;           // 33.5 MB
  bf16* xb   = (bf16*)ws;
  bf16* wb   = (bf16*)(ws + xbBytes);
  bf16* fb   = (bf16*)(ws + xbBytes + wbBytes);
  bf16* wcat = (bf16*)(ws + xbBytes + wbBytes + fBytes);  // 64 KB

  float* proposals = (float*)d_out;
  float* cls_out   = proposals + (size_t)B_ * AANCH * 4;
  float* ign       = cls_out   + (size_t)B_ * AANCH * 2;

  k_feat_to_bf16<<<dim3(4096), 256, 0, stream>>>(feat, xb);
  k_w_to_bf16<<<dim3((9 * 512 * 512) / 256), 256, 0, stream>>>(conv_w, wb);
  k_head_w<<<dim3(64 * 512 / 256), 256, 0, stream>>>(cls_w, reg_w, wcat);
  k_conv3x3<<<dim3(128, 4), 256, 0, stream>>>(xb, wb, conv_b, fb);
  k_head<<<dim3(256), 256, 0, stream>>>(fb, wcat, cls_b, reg_b, img,
                                        proposals, cls_out, ign);
}
